// ImpEntDecoder_46102178955800
// MI455X (gfx1250) — compile-verified, matrix operand order checked by reading, not run on hardware
//
#include <hip/hip_runtime.h>

// ---------------------------------------------------------------------------
// ImpEntDecoder: masked-causal conv entropy model on MI455X (gfx1250).
// Residual/final convs run as implicit GEMM on v_wmma_f32_16x16x32_bf16
// (f32 accumulate). Activations channels-last bf16, padded 144->160 so the
// K dimension is 5 chunks of 32. Weights pre-swizzled into the WMMA B-matrix
// lane layout and staged through LDS per tap -- via the gfx1250 async
// global->LDS path (ASYNCcnt) when the toolchain exposes it.
// ---------------------------------------------------------------------------

typedef __attribute__((ext_vector_type(16))) __bf16   v16bf;
typedef __attribute__((ext_vector_type(8)))  float    v8f;
typedef __attribute__((ext_vector_type(4)))  unsigned u32x4;

// async global->LDS builtin operand types (from hipcc diagnostic):
//   param0: v4i in AS1 ("__device__"), param1: v4i in AS3 ("__shared__")
typedef __attribute__((__vector_size__(16))) int vi4;
typedef __attribute__((address_space(1))) vi4 as1_vi4;
typedef __attribute__((address_space(3))) vi4 as3_vi4;

#define HH      256
#define WW      512
#define NPIX    (HH * WW)       // 131072
#define CPG     144
#define CPAD    160             // padded channels (5 chunks of 32)
#define KCH     5               // CPAD / 32
#define NCLS    49              // BIN + 1
#define CLSPAD  64
#define NTAP_C  13              // mask with center
#define NTAP_NC 12              // strictly causal mask
#define RES_TILES 9             // 144 / 16
#define FIN_TILES 4             // 64 / 16

#if defined(__gfx1250__) && __has_builtin(__builtin_amdgcn_global_load_async_to_lds_b128) && \
    __has_builtin(__builtin_amdgcn_s_wait_asynccnt)
#define USE_ASYNC_LDS 1
#else
#define USE_ASYNC_LDS 0
#endif

// tap -> (ky,kx) of the 5x5 kernel; dy = ky-2, dx = kx-2
__constant__ int TAPY_C[NTAP_C]  = {0,0,0,0,0, 1,1,1,1,1, 2,2,2};
__constant__ int TAPX_C[NTAP_C]  = {0,1,2,3,4, 0,1,2,3,4, 0,1,2};
__constant__ int TAPY_NC[NTAP_NC]= {0,0,0,0,0, 1,1,1,1,1, 2,2};
__constant__ int TAPX_NC[NTAP_NC]= {0,1,2,3,4, 0,1,2,3,4, 0,1};

__device__ __forceinline__ unsigned short f2bf(float f) {
    unsigned u = __float_as_uint(f);
    u += 0x7fffu + ((u >> 16) & 1u);       // round-to-nearest-even
    return (unsigned short)(u >> 16);
}
__device__ __forceinline__ float bf2f(unsigned short h) {
    return __uint_as_float(((unsigned)h) << 16);
}

// ---------------------------------------------------------------------------
// Weight pre-swizzle into WMMA B-fragment layout (32Kx16N bf16, wave32):
// lane = {half(K-block), col(N)}, dword v packs K pairs:
//   kbase = (v<4)?0:16 ; K0 = kbase + half*8 + 2*(v&3) ; K1 = K0+1
// Residual layout: [conv(10)][tap(13)][tile(9)][kchunk(5)][lane(32)][v(8)]
// ---------------------------------------------------------------------------
__global__ void prep_w_res(const float* __restrict__ wr, unsigned* __restrict__ wB) {
    long long idx = (long long)blockIdx.x * blockDim.x + threadIdx.x;
    const long long total = 10LL * NTAP_C * RES_TILES * KCH * 32 * 8;
    if (idx >= total) return;
    int v    = (int)(idx & 7);
    int lane = (int)((idx >> 3) & 31);
    long long r = idx >> 8;
    int kc   = (int)(r % KCH);       r /= KCH;
    int tile = (int)(r % RES_TILES); r /= RES_TILES;
    int tap  = (int)(r % NTAP_C);    r /= NTAP_C;
    int conv = (int)r;
    int half = lane >> 4, col = lane & 15;
    int K0 = ((v < 4) ? 0 : 16) + half * 8 + 2 * (v & 3);
    int ky = TAPY_C[tap], kx = TAPX_C[tap];
    int o  = tile * 16 + col;
    unsigned out = 0;
    for (int s = 0; s < 2; ++s) {
        int cin = kc * 32 + K0 + s;
        float w = 0.f;
        if (cin < CPG)
            w = wr[(((long long)conv * CPG + o) * CPG + cin) * 25 + ky * 5 + kx];
        out |= (unsigned)f2bf(w) << (16 * s);
    }
    wB[idx] = out;
}

// Final head layout: [tap(13)][tile(4)][kchunk(5)][lane(32)][v(8)], cout padded 49->64
__global__ void prep_w_fin(const float* __restrict__ wf, unsigned* __restrict__ wB) {
    long long idx = (long long)blockIdx.x * blockDim.x + threadIdx.x;
    const long long total = (long long)NTAP_C * FIN_TILES * KCH * 32 * 8;
    if (idx >= total) return;
    int v    = (int)(idx & 7);
    int lane = (int)((idx >> 3) & 31);
    long long r = idx >> 8;
    int kc   = (int)(r % KCH);       r /= KCH;
    int tile = (int)(r % FIN_TILES); r /= FIN_TILES;
    int tap  = (int)r;
    int half = lane >> 4, col = lane & 15;
    int K0 = ((v < 4) ? 0 : 16) + half * 8 + 2 * (v & 3);
    int ky = TAPY_C[tap], kx = TAPX_C[tap];
    int o  = tile * 16 + col;
    unsigned out = 0;
    for (int s = 0; s < 2; ++s) {
        int cin = kc * 32 + K0 + s;
        float w = 0.f;
        if (cin < CPG && o < NCLS)
            w = wf[((long long)o * CPG + cin) * 25 + ky * 5 + kx];
        out |= (unsigned)f2bf(w) << (16 * s);
    }
    wB[idx] = out;
}

// Zero the padded channel tail (144..159) of the three activation buffers.
__global__ void zero_pad(unsigned short* a, unsigned short* b, unsigned short* c) {
    long long i = (long long)blockIdx.x * blockDim.x + threadIdx.x;  // NPIX*16
    long long pix = i >> 4;
    int ch = CPG + (int)(i & 15);
    a[pix * CPAD + ch] = 0;
    b[pix * CPAD + ch] = 0;
    c[pix * CPAD + ch] = 0;
}

// First layer: 1 -> 144, strictly causal (12 taps). Tiny; direct VALU conv.
__global__ __launch_bounds__(256) void first_layer(
        const float* __restrict__ x, const float* __restrict__ w0,
        const float* __restrict__ b0, unsigned short* __restrict__ out) {
    __shared__ float lw[CPG * NTAP_NC];
    __shared__ float lb[CPG];
    for (int i = threadIdx.x; i < CPG * NTAP_NC; i += 256) {
        int c = i / NTAP_NC, t = i % NTAP_NC;
        lw[i] = w0[c * 25 + TAPY_NC[t] * 5 + TAPX_NC[t]];
    }
    for (int i = threadIdx.x; i < CPG; i += 256) lb[i] = b0[i];
    __syncthreads();
    int pix = blockIdx.x * 256 + threadIdx.x;          // grid == NPIX/256 exactly
    int y = pix / WW, xw = pix % WW;
    float xv[NTAP_NC];
    for (int t = 0; t < NTAP_NC; ++t) {
        int yy = y + TAPY_NC[t] - 2, xx = xw + TAPX_NC[t] - 2;
        xv[t] = (yy >= 0 && xx >= 0 && xx < WW) ? x[yy * WW + xx] : 0.f;
    }
    unsigned short* o = out + (long long)pix * CPAD;
    for (int c = 0; c < CPG; ++c) {
        float a = lb[c];
        const float* wc = &lw[c * NTAP_NC];
        for (int t = 0; t < NTAP_NC; ++t) a = fmaf(wc[t], xv[t], a);
        o[c] = f2bf(a > 0.f ? a : 0.f);
    }
}

// ---------------------------------------------------------------------------
// Implicit-GEMM masked conv via v_wmma_f32_16x16x32_bf16.
//   M=16 pixels (one row segment), N=16 couts per tile (wave holds NT tiles),
//   K = 5 chunks of 32 input channels, summed over 13 causal taps.
// MODE 0: out = relu(acc+bias) -> bf16
// MODE 1: out = resid + relu(acc+bias) -> bf16
// MODE 2: out = acc+bias -> f32 logits (stride CLSPAD)
// Template params keep all tile loops constant-trip so acc[] stays in VGPRs.
// ---------------------------------------------------------------------------
template <int NT, int CTOT, int MODE>
__global__ __launch_bounds__(128) void conv_wmma(
        const unsigned short* __restrict__ in,   // [NPIX][CPAD] bf16 bits
        const unsigned* __restrict__ wB,         // [tap][tile][kchunk][lane][8]
        const float* __restrict__ bias,
        const unsigned short* __restrict__ resid,
        unsigned short* __restrict__ out_bf,
        float* __restrict__ out_f32) {
    __shared__ unsigned ldsB[NT * KCH * 256];          // 46 KB (NT=9) / 20 KB (NT=4)

    const int lane = threadIdx.x & 31;
    const int wave = threadIdx.x >> 5;
    const int pixTile = blockIdx.x * 4 + wave;         // 8192 tiles total
    const int y  = pixTile >> 5;                       // 32 tiles per row
    const int x0 = (pixTile & 31) << 4;
    const int half = lane >> 4;
    const int m16  = lane & 15;                        // A-matrix row (pixel in tile)

    v8f acc[NT];
#pragma unroll
    for (int t = 0; t < NT; ++t)
        acc[t] = (v8f){0.f,0.f,0.f,0.f,0.f,0.f,0.f,0.f};

    constexpr int fragDw = NT * KCH * 256;

    for (int tap = 0; tap < NTAP_C; ++tap) {
        __syncthreads();
        {   // stage all B fragments for this tap into LDS
            const unsigned* src = wB + (long long)tap * fragDw;
#if USE_ASYNC_LDS
            // gfx1250 async global->LDS DMA, 16B granules, tracked by ASYNCcnt
            constexpr int granules = fragDw / 4;       // b128 units
            for (int g = threadIdx.x; g < granules; g += 128) {
                as1_vi4* gp = (as1_vi4*)(unsigned long long)
                                  (const void*)(src + g * 4);
                as3_vi4* lp = (as3_vi4*)(unsigned)
                                  (unsigned long long)(const void*)&ldsB[g * 4];
                __builtin_amdgcn_global_load_async_to_lds_b128(gp, lp, 0, 0);
            }
            __builtin_amdgcn_s_wait_asynccnt(0);
#else
#pragma unroll
            for (int i = 0; i < fragDw / 128; ++i)
                ldsB[i * 128 + threadIdx.x] = src[i * 128 + threadIdx.x];
#endif
        }
        __syncthreads();

        // prefetch next tap's fragments into L2 while we compute this one
        if (tap + 1 < NTAP_C) {
            const char* nxt = (const char*)(wB + (long long)(tap + 1) * fragDw);
            __builtin_prefetch(nxt + threadIdx.x * (fragDw * 4 / 128), 0, 1);
        }

        const int dy = TAPY_C[tap] - 2, dx = TAPX_C[tap] - 2;
        const int yy = y + dy;                         // dy<=0 so yy<HH always
        if (yy < 0) continue;                          // uniform across block
        const int px = x0 + dx + m16;
        const bool inb = (px >= 0) && (px < WW);
        const unsigned short* rowp =
            in + ((long long)yy * WW + px) * CPAD + half * 8;

#pragma unroll
        for (int kc = 0; kc < KCH; ++kc) {
            union { u32x4 q[2]; v16bf v; } a;
            if (inb) {
                const u32x4* p = (const u32x4*)(rowp + kc * 32);
                a.q[0] = p[0];     // K = kc*32 + half*8 + [0..7]
                a.q[1] = p[2];     // K = kc*32 + 16 + half*8 + [0..7]
            } else {
                a.q[0] = (u32x4){0u,0u,0u,0u};
                a.q[1] = (u32x4){0u,0u,0u,0u};
            }
#pragma unroll
            for (int t = 0; t < NT; ++t) {
                union { u32x4 q[2]; v16bf v; } b;
                const u32x4* bp = (const u32x4*)&ldsB[(t * KCH + kc) * 256 + lane * 8];
                b.q[0] = bp[0];
                b.q[1] = bp[1];
                acc[t] = __builtin_amdgcn_wmma_f32_16x16x32_bf16(
                    false, a.v, false, b.v, (short)0, acc[t], false, false);
            }
        }
    }

    // Epilogue. C/D layout: lane -> N (cout), VGPR r -> M (pixel), +8 for upper half.
    const int n = lane & 15;
#pragma unroll
    for (int t = 0; t < NT; ++t) {
        const int coutIdx = t * 16 + n;
        const float bv = (coutIdx < CTOT) ? bias[coutIdx] : 0.f;
#pragma unroll
        for (int r = 0; r < 8; ++r) {
            const int m = r + half * 8;
            const long long pix = (long long)y * WW + x0 + m;
            float vv = acc[t][r] + bv;
            if (MODE == 2) {
                out_f32[pix * CLSPAD + coutIdx] = vv;
            } else {
                float o = vv > 0.f ? vv : 0.f;
                if (MODE == 1) o += bf2f(resid[pix * CPAD + coutIdx]);
                out_bf[pix * CPAD + coutIdx] = f2bf(o);
            }
        }
    }
}

// Per-pixel softmax over 49 symbols; writes NCHW (1,49,256,512).
__global__ __launch_bounds__(256) void softmax_k(
        const float* __restrict__ logits, float* __restrict__ probs) {
    int pix = blockIdx.x * 256 + threadIdx.x;            // grid == NPIX/256
    const float* l = logits + (long long)pix * CLSPAD;
    float mx = l[0];
    for (int c = 1; c < NCLS; ++c) mx = fmaxf(mx, l[c]);
    float s = 0.f;
    for (int c = 0; c < NCLS; ++c) s += __expf(l[c] - mx);
    float inv = 1.f / s;
    for (int c = 0; c < NCLS; ++c)
        probs[(long long)c * NPIX + pix] = __expf(l[c] - mx) * inv;
}

// Importance code -> channel mask -> depth-to-space x2.  t: (48, 512, 1024)
__global__ __launch_bounds__(256) void tail_k(
        const float* __restrict__ x, float* __restrict__ t) {
    long long i = (long long)blockIdx.x * 256 + threadIdx.x;  // 48*512*1024
    int x2 = (int)(i % 1024);
    long long r = i / 1024;
    int y2 = (int)(r % 512);
    int c48 = (int)(r / 512);
    int h = y2 >> 1, a = y2 & 1, w = x2 >> 1, b = x2 & 1;
    int ch = c48 * 4 + a * 2 + b;
    float xv = x[h * WW + w];
    float tc = floorf((xv + 1.f) * 23.5f + 1e-5f);   // 1/SCALE = 47/2
    t[i] = ((float)ch < tc * 4.f) ? 1.f : 0.f;
}

// ---------------------------------------------------------------------------
extern "C" void kernel_launch(void* const* d_in, const int* in_sizes, int n_in,
                              void* d_out, int out_size, void* d_ws, size_t ws_size,
                              hipStream_t stream) {
    const float* x  = (const float*)d_in[0];
    const float* w0 = (const float*)d_in[1];
    const float* b0 = (const float*)d_in[2];
    const float* wr = (const float*)d_in[3];
    const float* br = (const float*)d_in[4];
    const float* wf = (const float*)d_in[5];
    const float* bfin = (const float*)d_in[6];

    float* probs = (float*)d_out;
    float* tout  = probs + (long long)NCLS * NPIX;

    // workspace carve-up
    unsigned short* actA = (unsigned short*)d_ws;                 // 3x bf16 act buffers
    unsigned short* actB = actA + (long long)NPIX * CPAD;
    unsigned short* actC = actB + (long long)NPIX * CPAD;
    float*    logits = (float*)(actC + (long long)NPIX * CPAD);   // [NPIX][64] f32
    unsigned* wBres  = (unsigned*)(logits + (long long)NPIX * CLSPAD);
    unsigned* wBfin  = wBres + 10LL * NTAP_C * RES_TILES * KCH * 256;

    const long long totRes = 10LL * NTAP_C * RES_TILES * KCH * 32 * 8;
    const long long totFin = (long long)NTAP_C * FIN_TILES * KCH * 32 * 8;
    prep_w_res<<<(int)((totRes + 255) / 256), 256, 0, stream>>>(wr, wBres);
    prep_w_fin<<<(int)((totFin + 255) / 256), 256, 0, stream>>>(wf, wBfin);
    zero_pad<<<NPIX * 16 / 256, 256, 0, stream>>>(actA, actB, actC);
    first_layer<<<NPIX / 256, 256, 0, stream>>>(x, w0, b0, actA);

    const int convGrid = NPIX / 16 / 4;   // 2048 blocks x 4 waves
    unsigned short* hbuf = actA;
    unsigned short* obuf = actC;
    const long long convStride = (long long)NTAP_C * RES_TILES * KCH * 256;
    for (int blk = 0; blk < 5; ++blk) {
        const unsigned* wB1 = wBres + (2LL * blk) * convStride;
        const unsigned* wB2 = wBres + (2LL * blk + 1) * convStride;
        conv_wmma<RES_TILES, CPG, 0><<<convGrid, 128, 0, stream>>>(
            hbuf, wB1, br + (2 * blk) * CPG, nullptr, actB, nullptr);
        conv_wmma<RES_TILES, CPG, 1><<<convGrid, 128, 0, stream>>>(
            actB, wB2, br + (2 * blk + 1) * CPG, hbuf, obuf, nullptr);
        unsigned short* tmp = hbuf; hbuf = obuf; obuf = tmp;
    }
    conv_wmma<FIN_TILES, NCLS, 2><<<convGrid, 128, 0, stream>>>(
        hbuf, wBfin, bfin, nullptr, nullptr, logits);

    softmax_k<<<NPIX / 256, 256, 0, stream>>>(logits, probs);
    tail_k<<<(int)((48LL * 512 * 1024) / 256), 256, 0, stream>>>(x, tout);
}
